// LSTM_86071144612213
// MI455X (gfx1250) — matrix operand order repaired and checked by reading, not compiled, not run on hardware
//
#include <hip/hip_runtime.h>
#include <hip/hip_bf16.h>
#include <math.h>

#define B_  256
#define T_  512
#define H_  1024
#define O_  10

typedef __bf16 bf16x8  __attribute__((ext_vector_type(8)));
typedef __bf16 v16bf   __attribute__((ext_vector_type(16)));
typedef float  v8f     __attribute__((ext_vector_type(8)));
typedef unsigned int v4u __attribute__((ext_vector_type(4)));
typedef int    v8i     __attribute__((ext_vector_type(8)));
typedef int    v4i     __attribute__((ext_vector_type(4)));

#if defined(__has_builtin)
#if __has_builtin(__builtin_amdgcn_tensor_load_to_lds) && __has_builtin(__builtin_amdgcn_s_wait_tensorcnt)
#define USE_TDM 1
#endif
#endif
#ifndef USE_TDM
#define USE_TDM 0
#endif

// ---------------------------------------------------------------------------
// Prep: convert + transpose the 4 recurrent weight matrices to bf16.
// WT layout: [gate][n][k]  (B-fragments contiguous along K)
// ---------------------------------------------------------------------------
__global__ __launch_bounds__(256) void lstm_prep_weights(
    const float* __restrict__ Wg, const float* __restrict__ Wi,
    const float* __restrict__ Wf, const float* __restrict__ Wo,
    __bf16* __restrict__ WT)
{
    int idx  = blockIdx.x * blockDim.x + threadIdx.x;   // 0 .. 4*1024*1024-1
    int n    = idx & (H_ - 1);
    int k    = (idx >> 10) & (H_ - 1);
    int gate = idx >> 20;
    const float* W = (gate == 0) ? Wg : (gate == 1) ? Wi : (gate == 2) ? Wf : Wo;
    WT[((size_t)gate << 20) + ((size_t)n << 10) + k] = (__bf16)W[((size_t)k << 10) + n];
}

__global__ __launch_bounds__(256) void lstm_zero_state(
    __bf16* __restrict__ h0, float* __restrict__ c)
{
    int i = blockIdx.x * blockDim.x + threadIdx.x;
    if (i < B_ * H_) {
        h0[i] = (__bf16)0.0f;
        c[i]  = 0.0f;
    }
}

#if USE_TDM
// Issue one TDM 2D tile load: 16 rows x 32 K-elements (2B each), row stride H_,
// from WT (viewed as 4096 x 1024 bf16 tensor) into LDS at lds_off.
// D# bitfields per CDNA5 ISA.
// clang-23 toolchain: 6-arg form (v4u, v8i, v4i, v4i, v8i, i32 cpol).
__device__ __forceinline__ void tdm_load_tile(unsigned lds_off, const __bf16* gptr)
{
    unsigned long long ga = (unsigned long long)(size_t)gptr;
    v4u g0;
    g0[0] = 1u;                                            // count=1 (valid), user mode
    g0[1] = lds_off;                                       // lds_addr (bytes)
    g0[2] = (unsigned)ga;                                  // global_addr[31:0]
    g0[3] = (unsigned)((ga >> 32) & 0x01FFFFFFu) | (2u << 30); // addr[56:32] | type=2
    v8i g1;
    g1[0] = (int)(1u << 16);                               // data_size=1 (2 bytes)
    g1[1] = (int)((H_ & 0xFFFF) << 16);                    // tensor_dim0[15:0]=1024
    g1[2] = (int)((H_ >> 16) | (((4 * H_) & 0xFFFF) << 16)); // dim0 hi | tensor_dim1=4096
    g1[3] = (int)(((4 * H_) >> 16) | (32u << 16));         // dim1 hi | tile_dim0=32
    g1[4] = 16;                                            // tile_dim1=16, tile_dim2=0
    g1[5] = H_;                                            // tensor_dim0_stride=1024
    g1[6] = 0;
    g1[7] = 0;
    v4i z4 = {0, 0, 0, 0};
    v8i z8 = {0, 0, 0, 0, 0, 0, 0, 0};
    __builtin_amdgcn_tensor_load_to_lds(g0, g1, z4, z4, z8, 0);
}
#endif

// ---------------------------------------------------------------------------
// One LSTM timestep.
// Block = one 16-wide N-panel (nt) x 8 M-tiles; all 8 waves share the B panel.
// Wave 0 drives a TDM double-buffer of the 4-gate B panel (4 KB/k-step) in LDS;
// each wave does 4x v_wmma_f32_16x16x32_bf16 per 32-K step, then the pointwise
// cell update.
// Grid: 128 blocks (64 nt x 2 m-halves) * 8 waves = 1024 waves.
// ---------------------------------------------------------------------------
__global__ __launch_bounds__(256) void lstm_step(
    const __bf16* __restrict__ h_in,    // [B][H] bf16 (previous h)
    __bf16*       __restrict__ h_out,   // [B][H] bf16 (new h)
    float*        __restrict__ c,       // [B][H] f32 cell state (read+write)
    float*        __restrict__ h_f32,   // [B][H] f32 (new h, for projection)
    const __bf16* __restrict__ WT,      // [4][H(n)][H(k)] bf16
    const float*  __restrict__ x,       // [B][T] f32
    const float* __restrict__ Wgx, const float* __restrict__ bg,
    const float* __restrict__ Wix, const float* __restrict__ bi,
    const float* __restrict__ Wfx, const float* __restrict__ bfv,
    const float* __restrict__ Wox, const float* __restrict__ bo,
    int t)
{
    const int wave  = threadIdx.x >> 5;
    const int lane  = threadIdx.x & 31;
    const int nt    = blockIdx.x >> 1;          // 64 N panels
    const int mhalf = blockIdx.x & 1;           // 2 halves of 16 M tiles
    const int m0    = (mhalf * 8 + wave) << 4;
    const int n0    = nt << 4;
    const int hl    = lane >> 4;
    const int l15   = lane & 15;

    // A operand (16-bit A 16x32 layout): lane l: row M=l, elems 0-7 = K 0..7 (+16*hl),
    // elems 8-15 = K 8..15 (+16*hl).
    const __bf16* Abase = h_in + (size_t)(m0 + l15) * H_;

    v8f acc[4] = { v8f{}, v8f{}, v8f{}, v8f{} };

#if USE_TDM
    // LDS double buffer: 2 x [4 gates][16 cols][32 K] bf16 = 2 x 4 KB.
    __shared__ bf16x8 smv[512];
    const unsigned lds_base = (unsigned)(size_t)&smv[0];

    if (wave == 0) {
#pragma unroll
        for (int g = 0; g < 4; ++g)
            tdm_load_tile(lds_base + (unsigned)(g << 10),
                          WT + ((size_t)g << 20) + ((size_t)n0 << 10));
    }

    for (int ks = 0; ks < 32; ++ks) {
        const int cur = ks & 1;
        const int k0  = ks << 5;

        // A fragment: issue early, overlaps the barrier.
        bf16x8 alo = *(const bf16x8*)(Abase + k0 + hl * 16);
        bf16x8 ahi = *(const bf16x8*)(Abase + k0 + hl * 16 + 8);

        if (wave == 0) __builtin_amdgcn_s_wait_tensorcnt(0);
        __syncthreads();   // buf[cur] ready; everyone done reading buf[cur^1]

        if (wave == 0 && ks + 1 < 32) {
            const int kn = k0 + 32;
#pragma unroll
            for (int g = 0; g < 4; ++g)
                tdm_load_tile(lds_base + (unsigned)(((cur ^ 1) << 12) + (g << 10)),
                              WT + ((size_t)g << 20) + ((size_t)n0 << 10) + kn);
        }

        union { v16bf v; bf16x8 h[2]; } ua;
        ua.h[0] = alo; ua.h[1] = ahi;

        // B fragment (16-bit B 32x16 layout): col N=l15, K = hl*16 .. +15,
        // LDS tile row (col) stride 64B -> base idx in bf16x8 units.
        const int bidx = (cur << 8) + (l15 << 2) + (hl << 1);
        bf16x8 b0l = smv[bidx +   0], b0h = smv[bidx +   1];
        bf16x8 b1l = smv[bidx +  64], b1h = smv[bidx +  65];
        bf16x8 b2l = smv[bidx + 128], b2h = smv[bidx + 129];
        bf16x8 b3l = smv[bidx + 192], b3h = smv[bidx + 193];
        union { v16bf v; bf16x8 h[2]; } u0, u1, u2, u3;
        u0.h[0] = b0l; u0.h[1] = b0h;
        u1.h[0] = b1l; u1.h[1] = b1h;
        u2.h[0] = b2l; u2.h[1] = b2h;
        u3.h[0] = b3l; u3.h[1] = b3h;

        acc[0] = __builtin_amdgcn_wmma_f32_16x16x32_bf16(false, ua.v, false, u0.v, (short)0, acc[0], false, false);
        acc[1] = __builtin_amdgcn_wmma_f32_16x16x32_bf16(false, ua.v, false, u1.v, (short)0, acc[1], false, false);
        acc[2] = __builtin_amdgcn_wmma_f32_16x16x32_bf16(false, ua.v, false, u2.v, (short)0, acc[2], false, false);
        acc[3] = __builtin_amdgcn_wmma_f32_16x16x32_bf16(false, ua.v, false, u3.v, (short)0, acc[3], false, false);
    }
#else
    // Fallback: direct global B loads (previous working path).
    const int col = n0 + l15;
    const __bf16* Bbase = WT + (size_t)col * H_ + (size_t)hl * 16;
    const size_t gstride = (size_t)H_ * H_;
    for (int k0 = 0; k0 < H_; k0 += 32) {
        bf16x8 alo = *(const bf16x8*)(Abase + k0 + hl * 16);
        bf16x8 ahi = *(const bf16x8*)(Abase + k0 + hl * 16 + 8);
        union { v16bf v; bf16x8 h[2]; } ua;
        ua.h[0] = alo; ua.h[1] = ahi;
#pragma unroll
        for (int g = 0; g < 4; ++g) {
            const __bf16* bp = Bbase + g * gstride + k0;
            bf16x8 blo = *(const bf16x8*)(bp);
            bf16x8 bhi = *(const bf16x8*)(bp + 8);
            union { v16bf v; bf16x8 h[2]; } ub;
            ub.h[0] = blo; ub.h[1] = bhi;
            acc[g] = __builtin_amdgcn_wmma_f32_16x16x32_bf16(
                false, ua.v, false, ub.v, (short)0, acc[g], false, false);
        }
    }
#endif

    // Epilogue: C/D layout: VGPR r -> row M = m0 + hl*8 + r, col N = n0 + l15.
    const int   n   = n0 + l15;
    const float wgx = Wgx[n], wix = Wix[n], wfx = Wfx[n], wox = Wox[n];
    const float bgv = bg[n],  biv = bi[n],  bfvv = bfv[n], bov = bo[n];
    const int   mbase = m0 + hl * 8;

#pragma unroll
    for (int r = 0; r < 8; ++r) {
        const int row = mbase + r;
        const float xv = x[(size_t)row * T_ + t];

        const float pg = acc[0][r] + xv * wgx + bgv;
        const float pi = acc[1][r] + xv * wix + biv;
        const float pf = acc[2][r] + xv * wfx + bfvv;
        const float po = acc[3][r] + xv * wox + bov;

        const float gg = tanhf(pg);
        const float ii = 1.0f / (1.0f + __expf(-pi));
        const float ff = 1.0f / (1.0f + __expf(-pf));
        const float oo = 1.0f / (1.0f + __expf(-po));

        const size_t idx = (size_t)row * H_ + n;
        const float cn = gg * ii + c[idx] * ff;
        c[idx] = cn;
        const float hn = tanhf(cn) * oo;
        h_f32[idx] = hn;
        h_out[idx] = (__bf16)hn;
    }
}

// ---------------------------------------------------------------------------
// Final projection: out[b][o] = h[b,:] @ W_ph[:,o] + b_p[o]  (256x10, tiny)
// ---------------------------------------------------------------------------
__global__ __launch_bounds__(256) void lstm_proj(
    const float* __restrict__ h, const float* __restrict__ Wp,
    const float* __restrict__ bp, float* __restrict__ out)
{
    int tid = blockIdx.x * blockDim.x + threadIdx.x;
    if (tid >= B_ * O_) return;
    int b = tid / O_, o = tid % O_;
    float s = bp[o];
    const float* hb = h + (size_t)b * H_;
    for (int k = 0; k < H_; ++k)
        s += hb[k] * Wp[k * O_ + o];
    out[tid] = s;
}

// ---------------------------------------------------------------------------
extern "C" void kernel_launch(void* const* d_in, const int* in_sizes, int n_in,
                              void* d_out, int out_size, void* d_ws, size_t ws_size,
                              hipStream_t stream) {
    const float* x    = (const float*)d_in[0];
    const float* Wgx  = (const float*)d_in[1];
    const float* Wgh  = (const float*)d_in[2];
    const float* bg   = (const float*)d_in[3];
    const float* Wix  = (const float*)d_in[4];
    const float* Wih  = (const float*)d_in[5];
    const float* bi   = (const float*)d_in[6];
    const float* Wfx  = (const float*)d_in[7];
    const float* Wfh  = (const float*)d_in[8];
    const float* bfv  = (const float*)d_in[9];
    const float* Wox  = (const float*)d_in[10];
    const float* Woh  = (const float*)d_in[11];
    const float* bo   = (const float*)d_in[12];
    const float* Wph  = (const float*)d_in[13];
    const float* bp   = (const float*)d_in[14];

    char* ws = (char*)d_ws;
    size_t off = 0;
    __bf16* WT   = (__bf16*)(ws + off); off += (size_t)4 * H_ * H_ * sizeof(__bf16); // 8 MB
    __bf16* hB0  = (__bf16*)(ws + off); off += (size_t)B_ * H_ * sizeof(__bf16);     // 512 KB
    __bf16* hB1  = (__bf16*)(ws + off); off += (size_t)B_ * H_ * sizeof(__bf16);     // 512 KB
    float*  cbuf = (float*) (ws + off); off += (size_t)B_ * H_ * sizeof(float);      // 1 MB
    float*  hF32 = (float*) (ws + off); off += (size_t)B_ * H_ * sizeof(float);      // 1 MB

    lstm_prep_weights<<<(4 * H_ * H_) / 256, 256, 0, stream>>>(Wgh, Wih, Wfh, Woh, WT);
    lstm_zero_state<<<(B_ * H_) / 256, 256, 0, stream>>>(hB0, cbuf);

    for (int t = 0; t < T_; ++t) {
        const __bf16* hin  = (t & 1) ? hB1 : hB0;
        __bf16*       hout = (t & 1) ? hB0 : hB1;
        lstm_step<<<128, 256, 0, stream>>>(hin, hout, cbuf, hF32, WT, x,
                                           Wgx, bg, Wix, bi, Wfx, bfv, Wox, bo, t);
    }

    lstm_proj<<<(B_ * O_ + 255) / 256, 256, 0, stream>>>(hF32, Wph, bp, (float*)d_out);
}
